// ModularRIN_43954695307352
// MI455X (gfx1250) — matrix operand order repaired and checked by reading, not compiled, not run on hardware
//
#include <hip/hip_runtime.h>
#include <hip/hip_bf16.h>

typedef __attribute__((ext_vector_type(16))) _Float16 v16h;
typedef __attribute__((ext_vector_type(8)))  _Float16 v8h;
typedef __attribute__((ext_vector_type(8)))  float    v8f;

#define PHI_F    1.61803398874989484820f
#define QSCALE   651.8986469044033f      /* 4096 / (2*pi) */
#define QINV     0.0015339807878856412f  /* (2*pi) / 4096 */

static __device__ __forceinline__ v16h combine8(v8h lo, v8h hi) {
    return __builtin_shufflevector(lo, hi, 0,1,2,3,4,5,6,7,8,9,10,11,12,13,14,15);
}

static __device__ __forceinline__ void lut_sincos(float theta, float& s, float& c) {
    float q = rintf(theta * QSCALE) * QINV;   // quantize to 4096-bin LUT angle
    s = __sinf(q);
    c = __cosf(q);
}

// ---------------------------------------------------------------------------
// Phase 1: elementwise quantized-trig recurrence over t = 0..511.
// Each thread owns one (b,d) element; h_real/h_imag chain is independent of
// the MLP layers (their result is dead until the final step).
// Grid: 384 blocks x 256 threads == 2048*48 elements exactly.
// ---------------------------------------------------------------------------
__global__ __launch_bounds__(256) void rin_scan_kernel(
    const int*   __restrict__ ids,   // [2048,512]
    const float* __restrict__ TE,    // [113,96]
    float*       __restrict__ h0out) // [2048,48]  c+s at t=511
{
    __shared__ float te[113 * 96];
    for (int i = threadIdx.x; i < 113 * 96; i += 256) te[i] = TE[i];
    __syncthreads();

    int gid = blockIdx.x * 256 + threadIdx.x;   // 0 .. 98303
    int b = gid / 48;
    int d = gid - b * 48;
    const int* idsb = ids + b * 512;

    float hr = 0.0f, hi = 0.0f;
    for (int t = 0; t < 512; ++t) {
        int tok = idsb[t];                       // uniform per row -> 1 req/wave
        float w  = te[tok * 96 + d];
        float bb = te[tok * 96 + 48 + d];
        float theta = (hr + hi) / (1.0f + fabsf(w)) + bb + (float)t * PHI_F;
        lut_sincos(theta, hi, hr);               // hi = sin, hr = cos
    }
    h0out[gid] = hr + hi;
}

// ---------------------------------------------------------------------------
// Phase 2+3: apply the L=2 MLP layers once (t_val = 511*phi) then the final
// vocab projection, all with v_wmma_f32_16x16x32_f16.
// Block = 128 threads (4 waves); each wave owns 16 rows of B. Grid = 32.
// ---------------------------------------------------------------------------
__global__ __launch_bounds__(128) void rin_layers_kernel(
    const float* __restrict__ h0,        // [2048,48]
    const float* __restrict__ layer_W,   // [2,96,48]
    const float* __restrict__ layer_b,   // [2,96]
    const float* __restrict__ proj_r,    // [2,48,96]
    const float* __restrict__ proj_i,    // [2,48,96]
    const float* __restrict__ out_proj,  // [113,48]
    float*       __restrict__ out)       // [2048,113]
{
    // f16 weight staging, K padded to 64 / concatenated to 192 (mult of 32)
    __shared__ __align__(16) _Float16 W16[2][96][64];   // B of GEMM1: W[n][k]
    __shared__ float                  biasS[2][96];
    __shared__ __align__(16) _Float16 PJ[2][48][192];   // [real | imag] per d-row
    __shared__ __align__(16) _Float16 OP[128][64];      // vocab proj, zero padded
    __shared__ __align__(16) _Float16 hA[4][16][64];    // per-wave A staging
    __shared__ __align__(16) _Float16 csA[4][16][192];  // per-wave [cos|sin] A

    const int tid = threadIdx.x;
    for (int i = tid; i < 2 * 96 * 48; i += 128) {
        int l = i / (96 * 48), rem = i % (96 * 48);
        W16[l][rem / 48][rem % 48] = (_Float16)layer_W[i];
    }
    for (int i = tid; i < 2 * 96 * 16; i += 128) {      // zero K pad 48..63
        int l = i / (96 * 16), rem = i % (96 * 16);
        W16[l][rem / 16][48 + rem % 16] = (_Float16)0.0f;
    }
    for (int i = tid; i < 2 * 96; i += 128) biasS[i / 96][i % 96] = layer_b[i];
    for (int i = tid; i < 2 * 48 * 96; i += 128) {
        int l = i / (48 * 96), rem = i % (48 * 96);
        int d = rem / 96, n = rem % 96;
        PJ[l][d][n]      = (_Float16)proj_r[i];
        PJ[l][d][96 + n] = (_Float16)proj_i[i];
    }
    for (int i = tid; i < 128 * 64; i += 128) {
        int v = i / 64, k = i % 64;
        OP[v][k] = (v < 113 && k < 48) ? (_Float16)out_proj[v * 48 + k] : (_Float16)0.0f;
    }
    __syncthreads();

    const int wave   = tid >> 5;
    const int lane   = tid & 31;
    const int laneLo = lane & 15;     // N (or M for A-frag rows)
    const int laneHi = lane >> 4;
    const int rowBase = blockIdx.x * 64 + wave * 16;
    const float t_val = 511.0f * PHI_F;

    _Float16* myhA = &hA[wave][0][0];
    _Float16* mycs = &csA[wave][0][0];
    for (int i = lane; i < 16 * 64; i += 32) myhA[i] = (_Float16)0.0f; // K pad

    // h state in native WMMA C layout: element (M = r + 8*laneHi, n = tile*16+laneLo)
    float h[3][8];
    #pragma unroll
    for (int tile = 0; tile < 3; ++tile)
        #pragma unroll
        for (int r = 0; r < 8; ++r)
            h[tile][r] = h0[(rowBase + r + 8 * laneHi) * 48 + tile * 16 + laneLo];

    #pragma unroll
    for (int l = 0; l < 2; ++l) {
        // ---- stage h as f16 A-matrix (wave-private LDS, no block barrier) ----
        #pragma unroll
        for (int tile = 0; tile < 3; ++tile)
            #pragma unroll
            for (int r = 0; r < 8; ++r)
                myhA[(r + 8 * laneHi) * 64 + tile * 16 + laneLo] = (_Float16)h[tile][r];
        asm volatile("s_wait_dscnt 0x0" ::: "memory");

        // A frags: lane row M=laneLo; K chunks [kb + 8*laneHi .. +7] and [+16]
        v16h aF[2];
        #pragma unroll
        for (int ks = 0; ks < 2; ++ks) {
            const _Float16* ap = myhA + laneLo * 64 + 32 * ks + (laneHi ? 8 : 0);
            aF[ks] = combine8(*(const v8h*)ap, *(const v8h*)(ap + 16));
        }

        // ---- GEMM1: th[16x96] = h @ W^T + bias + t_val; then LUT sin/cos ----
        #pragma unroll
        for (int nt = 0; nt < 6; ++nt) {
            v8f acc = {};
            #pragma unroll
            for (int ks = 0; ks < 2; ++ks) {
                const _Float16* bp = &W16[l][nt * 16 + laneLo][32 * ks + (laneHi ? 8 : 0)];
                v16h bF = combine8(*(const v8h*)bp, *(const v8h*)(bp + 16));
                acc = __builtin_amdgcn_wmma_f32_16x16x32_f16(false, aF[ks], false, bF,
                                                             (short)0, acc, false, false);
            }
            int n = nt * 16 + laneLo;
            float bn = biasS[l][n];
            #pragma unroll
            for (int r = 0; r < 8; ++r) {
                int M = r + 8 * laneHi;
                float s, c;
                lut_sincos(acc[r] + bn + t_val, s, c);
                mycs[M * 192 + n]      = (_Float16)c;   // [cos | sin] concat, K=192
                mycs[M * 192 + 96 + n] = (_Float16)s;
            }
        }
        asm volatile("s_wait_dscnt 0x0" ::: "memory");

        // ---- GEMM2: out[16x48] = [c|s] @ [proj_r|proj_i]^T ; h += silu(out) ----
        v16h a2[6];
        #pragma unroll
        for (int ks = 0; ks < 6; ++ks) {
            const _Float16* ap = mycs + laneLo * 192 + 32 * ks + (laneHi ? 8 : 0);
            a2[ks] = combine8(*(const v8h*)ap, *(const v8h*)(ap + 16));
        }
        #pragma unroll
        for (int dt = 0; dt < 3; ++dt) {
            v8f acc = {};
            #pragma unroll
            for (int ks = 0; ks < 6; ++ks) {
                const _Float16* bp = &PJ[l][dt * 16 + laneLo][32 * ks + (laneHi ? 8 : 0)];
                v16h bF = combine8(*(const v8h*)bp, *(const v8h*)(bp + 16));
                acc = __builtin_amdgcn_wmma_f32_16x16x32_f16(false, a2[ks], false, bF,
                                                             (short)0, acc, false, false);
            }
            #pragma unroll
            for (int r = 0; r < 8; ++r) {
                float o = acc[r];
                h[dt][r] += o / (1.0f + __expf(-o));   // silu
            }
        }
    }

    // ---- Final vocab projection: out = h @ out_proj^T, N padded 113 -> 128 ----
    #pragma unroll
    for (int tile = 0; tile < 3; ++tile)
        #pragma unroll
        for (int r = 0; r < 8; ++r)
            myhA[(r + 8 * laneHi) * 64 + tile * 16 + laneLo] = (_Float16)h[tile][r];
    asm volatile("s_wait_dscnt 0x0" ::: "memory");

    v16h aF[2];
    #pragma unroll
    for (int ks = 0; ks < 2; ++ks) {
        const _Float16* ap = myhA + laneLo * 64 + 32 * ks + (laneHi ? 8 : 0);
        aF[ks] = combine8(*(const v8h*)ap, *(const v8h*)(ap + 16));
    }
    #pragma unroll
    for (int nt = 0; nt < 8; ++nt) {
        v8f acc = {};
        #pragma unroll
        for (int ks = 0; ks < 2; ++ks) {
            const _Float16* bp = &OP[nt * 16 + laneLo][32 * ks + (laneHi ? 8 : 0)];
            v16h bF = combine8(*(const v8h*)bp, *(const v8h*)(bp + 16));
            acc = __builtin_amdgcn_wmma_f32_16x16x32_f16(false, aF[ks], false, bF,
                                                         (short)0, acc, false, false);
        }
        int v = nt * 16 + laneLo;
        if (v < 113) {
            #pragma unroll
            for (int r = 0; r < 8; ++r)
                out[(rowBase + r + 8 * laneHi) * 113 + v] = acc[r];
        }
    }
}

extern "C" void kernel_launch(void* const* d_in, const int* in_sizes, int n_in,
                              void* d_out, int out_size, void* d_ws, size_t ws_size,
                              hipStream_t stream) {
    const int*   ids  = (const int*)  d_in[0];  // [2048,512]
    const float* TE   = (const float*)d_in[1];  // [113,96]
    const float* lW   = (const float*)d_in[2];  // [2,96,48]
    const float* lB   = (const float*)d_in[3];  // [2,96]
    const float* pR   = (const float*)d_in[4];  // [2,48,96]
    const float* pI   = (const float*)d_in[5];  // [2,48,96]
    const float* oP   = (const float*)d_in[6];  // [113,48]
    float*       outp = (float*)d_out;          // [2048,113]

    float* h0 = (float*)d_ws;                   // [2048,48] scratch

    rin_scan_kernel<<<384, 256, 0, stream>>>(ids, TE, h0);
    rin_layers_kernel<<<32, 128, 0, stream>>>(h0, lW, lB, pR, pI, oP, outp);
}
